// GRU_AE_79834852098589
// MI455X (gfx1250) — compile-verified, matrix operand order checked by reading, not run on hardware
//
#include <hip/hip_runtime.h>
#include <hip/hip_bf16.h>

// ---------------------------------------------------------------------------
// CDNA5 (gfx1250) implementation: bf16 WMMA GEMM cores (f32 accumulate),
// fused GRU-step kernels (WMMA -> LDS -> gates), fused attention kernel.
// wave32 everywhere.  Fragment loads are branch-free and two-phase:
// batched float4 raw loads first (one clause, overlapped latency), then
// packed bf16 conversion.  Ragged-N B fragments CLAMP the row index (the
// garbage columns are discarded at the store guard).
// ---------------------------------------------------------------------------

typedef __bf16 v16bf __attribute__((ext_vector_type(16)));
typedef float  v8f   __attribute__((ext_vector_type(8)));
typedef float  v4f   __attribute__((ext_vector_type(4)));

#define BATCH 512
#define TT    32
#define SPOS  96          // A*T attention positions
#define NEGBIG (-1e30f)

// Raw (pre-conversion) 16x32 fragment: four 16B vector loads per lane.
// Lane l holds row (rowbase + (l&15)); K split per ISA 7.12.2:
//   elem e<8  -> k = kbase + (l>=16?8:0) + e
//   elem e>=8 -> k = kbase + 16 + (l>=16?8:0) + (e-8)
struct RawFrag { v4f q0, q1, q2, q3; };

template <bool CLAMP>
__device__ __forceinline__ RawFrag load_raw(const float* __restrict__ M, int ld,
                                            int rowbase, int kbase, int lane,
                                            int nrows)
{
    int r = rowbase + (lane & 15);
    if (CLAMP) r = (r < nrows) ? r : (nrows - 1);
    const float* p = M + (size_t)r * ld + kbase + ((lane >> 4) << 3);
    RawFrag f;
    f.q0 = *(const v4f*)(p);
    f.q1 = *(const v4f*)(p + 4);
    f.q2 = *(const v4f*)(p + 16);
    f.q3 = *(const v4f*)(p + 20);
    return f;
}

__device__ __forceinline__ v16bf cvt_frag(RawFrag q)
{
    v16bf f;
#pragma unroll
    for (int t = 0; t < 4; ++t) {
        f[t]      = (__bf16)q.q0[t];
        f[4 + t]  = (__bf16)q.q1[t];
        f[8 + t]  = (__bf16)q.q2[t];
        f[12 + t] = (__bf16)q.q3[t];
    }
    return f;
}

__device__ __forceinline__ v8f wmma_bf16(v16bf a, v16bf b, v8f c)
{
    return __builtin_amdgcn_wmma_f32_16x16x32_bf16(false, a, false, b,
                                                   (short)0, c, false, false);
}

// ---------------------------------------------------------------------------
// Generic GEMM: C[M,N] = act( scaleA * A[M,K] @ W[N,K]^T + bias )
// M must be a multiple of 16 (true at all call sites).  N, ldc arbitrary.
// act: 0 = none, 1 = leaky_relu(0.2)
// One 16x64 output strip per wave.  Per K-step: raw-load ALL five fragments
// (A + 4xB) as one clause, then convert, then 4 back-to-back WMMAs.
// ---------------------------------------------------------------------------
__global__ void __launch_bounds__(128)
k_gemm(const float* __restrict__ A, int lda, int K,
       const float* __restrict__ W, int ldw, int N, int M,
       const float* __restrict__ bias,
       float* __restrict__ C, long ldc, float scaleA, int act)
{
    int lane = threadIdx.x & 31;
    int wave = threadIdx.x >> 5;
    int groupsN = (N + 63) >> 6;              // 64-wide output strips
    int tilesM  = M >> 4;
    int tid = blockIdx.x * 4 + wave;
    if (tid >= tilesM * groupsN) return;
    int tm = tid / groupsN;
    int n0 = (tid - tm * groupsN) << 6;

    v8f acc0 = {0.f,0.f,0.f,0.f,0.f,0.f,0.f,0.f};
    v8f acc1 = acc0, acc2 = acc0, acc3 = acc0;

    const float* arow = A + (size_t)(tm * 16 + (lane & 15)) * lda;
    for (int kb = 0; kb < K; kb += 32) {
        // phase 1: issue all raw loads (compiler can clause these)
        RawFrag ra  = load_raw<false>(A, lda, tm * 16, kb, lane, 0);
        RawFrag rb0 = load_raw<true>(W, ldw, n0,      kb, lane, N);
        RawFrag rb1 = load_raw<true>(W, ldw, n0 + 16, kb, lane, N);
        RawFrag rb2 = load_raw<true>(W, ldw, n0 + 32, kb, lane, N);
        RawFrag rb3 = load_raw<true>(W, ldw, n0 + 48, kb, lane, N);
        if (kb + 32 < K)                       // stream A ahead (global_prefetch)
            __builtin_prefetch(arow + kb + 32, 0, 0);
        // phase 2: convert + multiply-accumulate
        v16bf a  = cvt_frag(ra);
        v16bf b0 = cvt_frag(rb0);
        v16bf b1 = cvt_frag(rb1);
        v16bf b2 = cvt_frag(rb2);
        v16bf b3 = cvt_frag(rb3);
        acc0 = wmma_bf16(a, b0, acc0);
        acc1 = wmma_bf16(a, b1, acc1);
        acc2 = wmma_bf16(a, b2, acc2);
        acc3 = wmma_bf16(a, b3, acc3);
    }

    int m0 = tm * 16 + ((lane >> 4) << 3);
    v8f accs[4] = {acc0, acc1, acc2, acc3};
#pragma unroll
    for (int j = 0; j < 4; ++j) {
        int n = n0 + j * 16 + (lane & 15);
        if (n < N) {
            float bz = bias ? bias[n] : 0.0f;
#pragma unroll
            for (int r = 0; r < 8; ++r) {
                float v = accs[j][r] * scaleA + bz;
                if (act == 1) v = (v > 0.f) ? v : 0.2f * v;
                C[(size_t)(m0 + r) * ldc + n] = v;
            }
        }
    }
}

// ---------------------------------------------------------------------------
// Fused GRU step.  One block = 16 batch rows, 12 waves (384 threads).
// gi = X @ Wih^T + bih (WMMA) if Wih != null, else gi = X (precomputed, ldx).
// gh = H @ Whh^T + bhh (WMMA).  Gates (PyTorch r,z,n order) -> Hout [B,64].
// Optional second output Ys (stride ldys).  In-place Hout==H is safe: all
// reads of H happen before the barrier or read-then-write by the same thread.
// ---------------------------------------------------------------------------
__global__ void __launch_bounds__(384)
k_gru_step(const float* __restrict__ X, int ldx, int K,
           const float* __restrict__ Wih, const float* __restrict__ bih,
           const float* __restrict__ H,
           const float* __restrict__ Whh, const float* __restrict__ bhh,
           float* __restrict__ Hout, float* __restrict__ Ys, long ldys)
{
    __shared__ float sgi[16 * 192];
    __shared__ float sgh[16 * 192];
    int lane = threadIdx.x & 31;
    int wave = threadIdx.x >> 5;          // 0..11 -> 12 column tiles of 192
    int rowbase = blockIdx.x * 16;

    {   // gh tile, K = 64 fully unrolled: all raw loads first
        RawFrag ra0 = load_raw<false>(H, 64, rowbase, 0, lane, 0);
        RawFrag rb0 = load_raw<false>(Whh, 64, wave * 16, 0, lane, 0);
        RawFrag ra1 = load_raw<false>(H, 64, rowbase, 32, lane, 0);
        RawFrag rb1 = load_raw<false>(Whh, 64, wave * 16, 32, lane, 0);
        v8f acc = {0.f,0.f,0.f,0.f,0.f,0.f,0.f,0.f};
        acc = wmma_bf16(cvt_frag(ra0), cvt_frag(rb0), acc);
        acc = wmma_bf16(cvt_frag(ra1), cvt_frag(rb1), acc);
        int n  = wave * 16 + (lane & 15);
        int m0 = (lane >> 4) << 3;
#pragma unroll
        for (int r = 0; r < 8; ++r) sgh[(m0 + r) * 192 + n] = acc[r] + bhh[n];
    }
    if (Wih) {   // gi tile via WMMA
        v8f acc = {0.f,0.f,0.f,0.f,0.f,0.f,0.f,0.f};
        for (int kb = 0; kb < K; kb += 32) {
            RawFrag ra = load_raw<false>(X, ldx, rowbase, kb, lane, 0);
            RawFrag rb = load_raw<false>(Wih, K, wave * 16, kb, lane, 0);
            acc = wmma_bf16(cvt_frag(ra), cvt_frag(rb), acc);
        }
        int n  = wave * 16 + (lane & 15);
        int m0 = (lane >> 4) << 3;
#pragma unroll
        for (int r = 0; r < 8; ++r) sgi[(m0 + r) * 192 + n] = acc[r] + bih[n];
    } else {     // gi precomputed (includes bias from the big GEMM)
        for (int i = threadIdx.x; i < 16 * 192; i += 384) {
            int r = i / 192, c = i - r * 192;
            sgi[i] = X[(size_t)(rowbase + r) * ldx + c];
        }
    }
    __syncthreads();

    for (int i = threadIdx.x; i < 16 * 64; i += 384) {
        int r = i >> 6, c = i & 63;
        float ir = sgi[r * 192 + c], iz = sgi[r * 192 + 64 + c], in = sgi[r * 192 + 128 + c];
        float hr = sgh[r * 192 + c], hz = sgh[r * 192 + 64 + c], hn = sgh[r * 192 + 128 + c];
        float rg = 1.f / (1.f + __expf(-(ir + hr)));
        float zg = 1.f / (1.f + __expf(-(iz + hz)));
        float ng = tanhf(in + rg * hn);
        float hp = H[(size_t)(rowbase + r) * 64 + c];
        float ho = (1.f - zg) * ng + zg * hp;
        Hout[(size_t)(rowbase + r) * 64 + c] = ho;
        if (Ys) Ys[(size_t)(rowbase + r) * ldys + c] = ho;
    }
}

// ---------------------------------------------------------------------------
// Fused decoder attention step.  One block = 16 batch rows, 256 threads.
// sw = s @ W_s^T (WMMA, waves 0..3) -> LDS; att[b,sp] = sum_d tanh(score_e+sw)*v
// masked softmax over 96 positions; c = pr @ enc_output.  Writes
// ginput[b] = [emb, c] (192) and dcat[b][64:256] = [c, emb] for [h,c,emb].
// ---------------------------------------------------------------------------
__global__ void __launch_bounds__(256)
k_attn(const float* __restrict__ S, const float* __restrict__ scoreE,
       const float* __restrict__ encOut, const unsigned char* __restrict__ mask,
       const float* __restrict__ attnW, const float* __restrict__ vvec,
       const float* __restrict__ embD, const int* __restrict__ Xi, int t,
       float* __restrict__ ginput, float* __restrict__ dcat)
{
    __shared__ float ssw[16 * 64];
    __shared__ float satt[16 * SPOS];
    __shared__ float sv[64];
    int lane = threadIdx.x & 31;
    int wave = threadIdx.x >> 5;
    int rowbase = blockIdx.x * 16;

    if (threadIdx.x < 64) sv[threadIdx.x] = vvec[threadIdx.x];
    if (wave < 4) {        // sw = s @ W_s^T   (W_s = attnW[:, :64], ld 192)
        RawFrag ra0 = load_raw<false>(S, 64, rowbase, 0, lane, 0);
        RawFrag rb0 = load_raw<false>(attnW, 192, wave * 16, 0, lane, 0);
        RawFrag ra1 = load_raw<false>(S, 64, rowbase, 32, lane, 0);
        RawFrag rb1 = load_raw<false>(attnW, 192, wave * 16, 32, lane, 0);
        v8f acc = {0.f,0.f,0.f,0.f,0.f,0.f,0.f,0.f};
        acc = wmma_bf16(cvt_frag(ra0), cvt_frag(rb0), acc);
        acc = wmma_bf16(cvt_frag(ra1), cvt_frag(rb1), acc);
        int n  = wave * 16 + (lane & 15);
        int m0 = (lane >> 4) << 3;
#pragma unroll
        for (int r = 0; r < 8; ++r) ssw[(m0 + r) * 64 + n] = acc[r];
    }
    __syncthreads();

    for (int i = threadIdx.x; i < 16 * SPOS; i += 256) {
        int r = i / SPOS, sp = i - r * SPOS;
        int b = rowbase + r;
        float a;
        if (mask[b * TT + (sp & 31)]) {       // mask tiled over the 3 attrs
            a = NEGBIG;
        } else {
            const float* se = scoreE + ((size_t)sp * BATCH + b) * 64;
            float sum = 0.f;
#pragma unroll 8
            for (int d = 0; d < 64; ++d)
                sum += tanhf(se[d] + ssw[r * 64 + d]) * sv[d];
            a = sum;
        }
        satt[r * SPOS + sp] = a;
    }
    __syncthreads();

    if (threadIdx.x < 16) {                   // softmax per row
        int r = threadIdx.x;
        float mx = NEGBIG;
        for (int sp = 0; sp < SPOS; ++sp) mx = fmaxf(mx, satt[r * SPOS + sp]);
        float sum = 0.f;
        for (int sp = 0; sp < SPOS; ++sp) {
            float e = __expf(satt[r * SPOS + sp] - mx);
            satt[r * SPOS + sp] = e;
            sum += e;
        }
        float inv = 1.f / sum;
        for (int sp = 0; sp < SPOS; ++sp) satt[r * SPOS + sp] *= inv;
    }
    __syncthreads();

    for (int i = threadIdx.x; i < 16 * 128; i += 256) {   // context c
        int r = i >> 7, e = i & 127;
        int b = rowbase + r;
        float sum = 0.f;
        for (int sp = 0; sp < SPOS; ++sp)
            sum += satt[r * SPOS + sp] * encOut[((size_t)sp * BATCH + b) * 128 + e];
        ginput[(size_t)b * 192 + 64 + e] = sum;
        dcat[(size_t)b * 256 + 64 + e]   = sum;
    }
    for (int i = threadIdx.x; i < 16 * 64; i += 256) {    // prev-token embedding
        int r = i >> 6, c = i & 63;
        int b = rowbase + r;
        int xp = Xi[b * TT + t];
        float ev = embD[(size_t)xp * 64 + c];
        ginput[(size_t)b * 192 + c]      = ev;
        dcat[(size_t)b * 256 + 192 + c]  = ev;
    }
}

// ---------------------------------------------------------------------------
// Small helper kernels
// ---------------------------------------------------------------------------
__global__ void k_embed(const int* __restrict__ X, const float* __restrict__ emb,
                        float* __restrict__ xa)
{
    int i = blockIdx.x * blockDim.x + threadIdx.x;       // T*B*64
    if (i >= TT * BATCH * 64) return;
    int c = i & 63, tb = i >> 6;
    int b = tb & (BATCH - 1), t = tb >> 9;
    xa[(size_t)tb * 128 + c] = emb[(size_t)X[b * TT + t] * 64 + c];
}

__global__ void k_out0(const int* __restrict__ X, float* __restrict__ out, int V)
{
    int i = blockIdx.x * blockDim.x + threadIdx.x;       // B*V
    if (i >= BATCH * V) return;
    int b = i / V, c = i - b * V;
    out[(size_t)b * TT * V + c] = (c == X[0]) ? 1.0f : 0.0f;
}

__global__ void k_zero(float* __restrict__ p, int n)
{
    int i = blockIdx.x * blockDim.x + threadIdx.x;
    if (i < n) p[i] = 0.f;
}

__global__ void k_concat_hm(const float* __restrict__ hf, const float* __restrict__ hb,
                            float* __restrict__ hm)
{
    int i = blockIdx.x * blockDim.x + threadIdx.x;       // B*64
    if (i >= BATCH * 64) return;
    int b = i >> 6, c = i & 63;
    hm[(size_t)b * 128 + c]      = hf[i];
    hm[(size_t)b * 128 + 64 + c] = hb[i];
}

// ---------------------------------------------------------------------------
// Host orchestration
// ---------------------------------------------------------------------------
extern "C" void kernel_launch(void* const* d_in, const int* in_sizes, int n_in,
                              void* d_out, int out_size, void* d_ws, size_t ws_size,
                              hipStream_t stream)
{
    (void)in_sizes; (void)n_in; (void)out_size; (void)ws_size;
    static const int VDIMS[3] = {201, 51, 11};

    auto F = [&](int i) { return (const float*)d_in[i]; };
    const unsigned char* maskp = (const unsigned char*)d_in[3];

    // Workspace layout (floats)
    float* ws = (float*)d_ws;
    float* xa    = ws;                       // [T*B,128] layer input
    float* xb    = ws + 2097152;             // [T*B,128] layer output
    float* gi    = ws + 4194304;             // [T*B,192] gate preacts
    float* enc   = ws + 7340032;             // [96*B,128] enc_output
    float* sce   = ws + 13631488;            // [96*B,64] score_e (per attr)
    float* hf    = ws + 16777216;            // [B,64]
    float* hb    = ws + 16809984;            // [B,64]
    float* hm    = ws + 16842752;            // [B,128]
    float* s0    = ws + 16908288;            // [3][B,64] decoder states
    float* h1    = ws + 17006592;            // [B,64]
    float* gin   = ws + 17039360;            // [B,192] decoder GRU input
    float* dcat  = ws + 17137664;            // [B,256] [h,c,emb]

    auto gemm = [&](const float* A, int lda, int K, const float* W, int ldw,
                    int N, int M, const float* bias, float* C, long ldc,
                    float scale, int act) {
        int tiles = (M / 16) * ((N + 63) / 64);
        k_gemm<<<(tiles + 3) / 4, 128, 0, stream>>>(A, lda, K, W, ldw, N, M,
                                                    bias, C, ldc, scale, act);
    };
    auto zero = [&](float* p, int n) {
        k_zero<<<(n + 255) / 256, 256, 0, stream>>>(p, n);
    };

    // ---------------- Encoders ----------------
    for (int i = 0; i < 3; ++i) {
        const int ei = 43 + 19 * i;          // enc param base (sorted pytree)
        const int* Xi = (const int*)d_in[i];

        k_embed<<<(TT * BATCH * 64 + 255) / 256, 256, 0, stream>>>(Xi, F(ei + 0), xa);

        const float* cur = xa;
        for (int l = 0; l < 2; ++l) {
            int K = (l == 0) ? 64 : 128;
            // layer-l params: bwd block first (sorted), order Whh,Wih,bhh,bih
            int pb = ei + 3 + l * 8;         // bwd
            int pf = pb + 4;                 // fwd
            const float *WhhB = F(pb), *WihB = F(pb + 1), *bhhB = F(pb + 2), *bihB = F(pb + 3);
            const float *WhhF = F(pf), *WihF = F(pf + 1), *bhhF = F(pf + 2), *bihF = F(pf + 3);

            // forward direction
            gemm(cur, 128, K, WihF, K, 192, TT * BATCH, bihF, gi, 192, 1.f, 0);
            zero(hf, BATCH * 64);
            for (int t = 0; t < TT; ++t) {
                float* ys = (l == 0)
                    ? xb + (size_t)t * BATCH * 128
                    : enc + ((size_t)(i * TT + t)) * BATCH * 128;
                k_gru_step<<<BATCH / 16, 384, 0, stream>>>(
                    gi + (size_t)t * BATCH * 192, 192, 0, nullptr, nullptr,
                    hf, WhhF, bhhF, hf, ys, 128);
            }
            // backward direction (scan reversed input, emit at original t)
            gemm(cur, 128, K, WihB, K, 192, TT * BATCH, bihB, gi, 192, 1.f, 0);
            zero(hb, BATCH * 64);
            for (int t = TT - 1; t >= 0; --t) {
                float* ys = (l == 0)
                    ? xb + (size_t)t * BATCH * 128 + 64
                    : enc + ((size_t)(i * TT + t)) * BATCH * 128 + 64;
                k_gru_step<<<BATCH / 16, 384, 0, stream>>>(
                    gi + (size_t)t * BATCH * 192, 192, 0, nullptr, nullptr,
                    hb, WhhB, bhhB, hb, ys, 128);
            }
            cur = xb;
        }
        // s0 = leaky_relu( (concat(hf,hb)/sqrt(1+eps)) @ fc_W^T + fc_b )
        k_concat_hm<<<(BATCH * 64 + 255) / 256, 256, 0, stream>>>(hf, hb, hm);
        gemm(hm, 128, 128, F(ei + 1), 128, 64, BATCH, F(ei + 2),
             s0 + (size_t)i * BATCH * 64, 64, 0.99999500003749968f, 1);
    }

    // ---------------- Decoders ----------------
    float* outbase = (float*)d_out;
    size_t ooff = 0;
    for (int i = 0; i < 3; ++i) {
        const int di = 4 + 13 * i;           // dec param base
        const int V = VDIMS[i];
        const int* Xi = (const int*)d_in[i];
        const float* attnW = F(di + 0);
        const float* embD  = F(di + 1);
        const float *Whh0 = F(di + 2), *Wih0 = F(di + 3), *bhh0 = F(di + 4), *bih0 = F(di + 5);
        const float *Whh1 = F(di + 6), *Wih1 = F(di + 7), *bhh1 = F(di + 8), *bih1 = F(di + 9);
        const float *outW = F(di + 10), *outb = F(di + 11), *vv = F(di + 12);
        float* outp = outbase + ooff;

        // score_e = enc_output @ W_e^T  (W_e = attnW[:, 64:192], ld 192)
        gemm(enc, 128, 128, attnW + 64, 192, 64, SPOS * BATCH, nullptr, sce, 64, 1.f, 0);

        // outputs[:,0,:] = one-hot(X[0,0])
        k_out0<<<(BATCH * V + 255) / 256, 256, 0, stream>>>(Xi, outp, V);

        float* s = s0 + (size_t)i * BATCH * 64;   // in-place state update is safe
        for (int t = 0; t < TT - 1; ++t) {
            k_attn<<<BATCH / 16, 256, 0, stream>>>(s, sce, enc, maskp, attnW, vv,
                                                   embD, Xi, t, gin, dcat);
            // decoder GRU layer 0: x=[emb,c] (192), hidden = s -> h1
            k_gru_step<<<BATCH / 16, 384, 0, stream>>>(gin, 192, 192, Wih0, bih0,
                                                       s, Whh0, bhh0, h1, nullptr, 0);
            // decoder GRU layer 1: x=h1, hidden = s -> s (also -> dcat[:, 0:64])
            k_gru_step<<<BATCH / 16, 384, 0, stream>>>(h1, 64, 64, Wih1, bih1,
                                                       s, Whh1, bhh1, s, dcat, 256);
            // pred = [h,c,emb] @ out_W^T + out_b  -> out[:, t+1, :]
            gemm(dcat, 256, 256, outW, 256, V, BATCH, outb,
                 outp + (size_t)(t + 1) * V, (long)TT * V, 1.f, 0);
        }
        ooff += (size_t)BATCH * TT * V;
    }
}
